// SimpleModel03_5755256176696
// MI455X (gfx1250) — compile-verified, matrix operand order checked by reading, not run on hardware
//
#include <hip/hip_runtime.h>
#include <hip/hip_bf16.h>

typedef _Float16 h8  __attribute__((ext_vector_type(8)));
typedef _Float16 v16h __attribute__((ext_vector_type(16)));
typedef float    v8f  __attribute__((ext_vector_type(8)));

#define DH 128   // hidden width (d_in == d_h == 128)

// ---------------------------------------------------------------------------
// Degree / dinv
// ---------------------------------------------------------------------------
__global__ void deg_init_kernel(float* __restrict__ deg, int n) {
    int i = blockIdx.x * blockDim.x + threadIdx.x;
    if (i < n) deg[i] = 1.0f;               // self-loop contributes 1
}

__global__ void deg_acc_kernel(const int* __restrict__ dst, float* __restrict__ deg, int E) {
    int e = blockIdx.x * blockDim.x + threadIdx.x;
    if (e < E) unsafeAtomicAdd(&deg[dst[e]], 1.0f);
}

__global__ void dinv_kernel(float* __restrict__ deg, int n) {
    int i = blockIdx.x * blockDim.x + threadIdx.x;
    if (i < n) {
        float d = deg[i];
        deg[i] = (d > 0.0f) ? rsqrtf(d) : 0.0f;
    }
}

// ---------------------------------------------------------------------------
// Conversions
// ---------------------------------------------------------------------------
// W is [K=128][N=128] row-major fp32; produce W^T f16: wt[n][k]  (K-contiguous
// per output feature => matches WMMA B-fragment per-lane layout).
__global__ void cvt_wT_kernel(const float* __restrict__ W, _Float16* __restrict__ wt) {
    int i = blockIdx.x * blockDim.x + threadIdx.x;   // flat index into wt
    if (i >= DH * DH) return;
    int n = i >> 7;          // output feature (row of wt)
    int k = i & 127;         // input feature
    wt[i] = (_Float16)W[k * DH + n];
}

__global__ void cvt_f16_kernel(const float* __restrict__ x, _Float16* __restrict__ xh, int total) {
    int i = blockIdx.x * blockDim.x + threadIdx.x;
    if (i < total) xh[i] = (_Float16)x[i];
}

__global__ void relu_cvt_kernel(const float* __restrict__ a, _Float16* __restrict__ xh, int total) {
    int i = blockIdx.x * blockDim.x + threadIdx.x;
    if (i < total) xh[i] = (_Float16)fmaxf(a[i], 0.0f);
}

// ---------------------------------------------------------------------------
// WMMA GEMM: C[M x 128] = A[M x 128](f16 row-major) * W(via W^T f16) ; f32 out
// One wave per 16-row M-tile; 8 N-tiles of 16; K-loop 4 x 32.
// ---------------------------------------------------------------------------
__global__ void __launch_bounds__(256)
gemm_wmma_kernel(const _Float16* __restrict__ A, const _Float16* __restrict__ WT,
                 float* __restrict__ C, int M) {
    __shared__ _Float16 ldsW[DH * DH];      // 32 KB: W^T, row = output feature
    {
        const uint4* s = (const uint4*)WT;
        uint4* d = (uint4*)ldsW;
        const int n16 = DH * DH * 2 / 16;   // 2048 x 16B
        for (int i = threadIdx.x; i < n16; i += blockDim.x) d[i] = s[i];
    }
    __syncthreads();

    const int wave  = threadIdx.x >> 5;
    const int lane  = threadIdx.x & 31;
    const int mtile = blockIdx.x * 8 + wave;
    if (mtile * 16 >= M) return;            // wave-uniform: EXEC stays all-ones

    const int mbase = mtile * 16;
    const int lrow  = lane & 15;            // M (A) or N (B) index within tile
    const int lhi   = lane >> 4;            // K-half selector
    const int kh    = lhi * 8;

    v8f acc[8] = {};

    const _Float16* arow = A + (size_t)(mbase + lrow) * DH;

#pragma unroll
    for (int ks = 0; ks < 4; ++ks) {
        const int kb = ks * 32;
        // A fragment: 16x32 f16, per-ISA layout (two 8-half chunks per lane)
        h8 alo = *(const h8*)(arow + kb + kh);
        h8 ahi = *(const h8*)(arow + kb + 16 + kh);
        v16h afrag;
#pragma unroll
        for (int i = 0; i < 8; ++i) { afrag[i] = alo[i]; afrag[8 + i] = ahi[i]; }

#pragma unroll
        for (int nt = 0; nt < 8; ++nt) {
            const _Float16* bcol = ldsW + (size_t)(nt * 16 + lrow) * DH;
            h8 blo = *(const h8*)(bcol + kb + kh);
            h8 bhi = *(const h8*)(bcol + kb + 16 + kh);
            v16h bfrag;
#pragma unroll
            for (int i = 0; i < 8; ++i) { bfrag[i] = blo[i]; bfrag[8 + i] = bhi[i]; }

            acc[nt] = __builtin_amdgcn_wmma_f32_16x16x32_f16(
                false, afrag, false, bfrag, (short)0, acc[nt], false, false);
        }
    }

    // C/D layout: VGPR r -> M = r + 8*(lane>>4); N = lane&15
#pragma unroll
    for (int nt = 0; nt < 8; ++nt) {
#pragma unroll
        for (int r = 0; r < 8; ++r) {
            const int m = mbase + r + lhi * 8;
            const int n = nt * 16 + lrow;
            C[(size_t)m * DH + n] = acc[nt][r];
        }
    }
}

// ---------------------------------------------------------------------------
// Aggregation: out = b + dinv^2 * h (self loop), then atomic edge scatter
// ---------------------------------------------------------------------------
__global__ void agg_init_kernel(const float* __restrict__ h, const float* __restrict__ dinv,
                                const float* __restrict__ b, float* __restrict__ agg, int total) {
    int i = blockIdx.x * blockDim.x + threadIdx.x;
    if (i >= total) return;
    int f = i & 127;
    int n = i >> 7;
    float d = dinv[n];
    agg[i] = b[f] + h[i] * d * d;
}

// One wave per edge: 32 lanes x float4 = 512B gather + 512B atomic scatter.
__global__ void __launch_bounds__(256)
agg_edges_kernel(const float* __restrict__ h, const float* __restrict__ dinv,
                 const int* __restrict__ src, const int* __restrict__ dst,
                 float* __restrict__ agg, int E) {
    int wid  = (int)((blockIdx.x * blockDim.x + threadIdx.x) >> 5);
    int lane = threadIdx.x & 31;
    if (wid >= E) return;
    int s = src[wid];
    int d = dst[wid];
    float norm = dinv[s] * dinv[d];
    float4 v = *(const float4*)(h + (size_t)s * DH + lane * 4);
    float* o = agg + (size_t)d * DH + lane * 4;
    unsafeAtomicAdd(o + 0, v.x * norm);
    unsafeAtomicAdd(o + 1, v.y * norm);
    unsafeAtomicAdd(o + 2, v.z * norm);
    unsafeAtomicAdd(o + 3, v.w * norm);
}

// ---------------------------------------------------------------------------
// Head: logits = relu(h2) @ W3 + b3 ; log_softmax over 2 classes
// ---------------------------------------------------------------------------
__global__ void head_kernel(const float* __restrict__ h, const float* __restrict__ W3,
                            const float* __restrict__ b3, float* __restrict__ out, int n) {
    int i = blockIdx.x * blockDim.x + threadIdx.x;
    if (i >= n) return;
    const float4* row = (const float4*)(h + (size_t)i * DH);
    float l0 = 0.0f, l1 = 0.0f;
#pragma unroll
    for (int k4 = 0; k4 < 32; ++k4) {
        float4 v = row[k4];
        float r0 = fmaxf(v.x, 0.0f), r1 = fmaxf(v.y, 0.0f);
        float r2 = fmaxf(v.z, 0.0f), r3 = fmaxf(v.w, 0.0f);
        int k = k4 * 4;
        l0 += r0 * W3[(k + 0) * 2 + 0] + r1 * W3[(k + 1) * 2 + 0]
            + r2 * W3[(k + 2) * 2 + 0] + r3 * W3[(k + 3) * 2 + 0];
        l1 += r0 * W3[(k + 0) * 2 + 1] + r1 * W3[(k + 1) * 2 + 1]
            + r2 * W3[(k + 2) * 2 + 1] + r3 * W3[(k + 3) * 2 + 1];
    }
    l0 += b3[0];
    l1 += b3[1];
    float m   = fmaxf(l0, l1);
    float lse = m + logf(expf(l0 - m) + expf(l1 - m));
    out[2 * i + 0] = l0 - lse;
    out[2 * i + 1] = l1 - lse;
}

// ---------------------------------------------------------------------------
extern "C" void kernel_launch(void* const* d_in, const int* in_sizes, int n_in,
                              void* d_out, int out_size, void* d_ws, size_t ws_size,
                              hipStream_t stream) {
    const float* x  = (const float*)d_in[0];
    const int*   ei = (const int*)d_in[1];
    const float* W1 = (const float*)d_in[2];
    const float* b1 = (const float*)d_in[3];
    const float* W2 = (const float*)d_in[4];
    const float* b2 = (const float*)d_in[5];
    const float* W3 = (const float*)d_in[6];
    const float* b3 = (const float*)d_in[7];
    float* out = (float*)d_out;

    const int NN = in_sizes[0] / DH;     // 100000 nodes
    const int E  = in_sizes[1] / 2;      // 1.6M edges
    const int* src = ei;
    const int* dst = ei + E;
    const int total = NN * DH;

    // workspace carve-out (256B aligned)
    char* w = (char*)d_ws;
    size_t off = 0;
    auto carve = [&](size_t bytes) -> void* {
        void* p = w + off;
        off = (off + bytes + 255) & ~(size_t)255;
        return p;
    };
    float*     dinv = (float*)    carve((size_t)NN * 4);
    _Float16*  w1t  = (_Float16*) carve((size_t)DH * DH * 2);
    _Float16*  w2t  = (_Float16*) carve((size_t)DH * DH * 2);
    _Float16*  hf16 = (_Float16*) carve((size_t)total * 2);  // x / relu(h) in f16
    float*     hbuf = (float*)    carve((size_t)total * 4);  // GEMM output
    float*     abuf = (float*)    carve((size_t)total * 4);  // aggregation output

    const int T = 256;
    dim3 blk(T);
    auto g1 = [&](int n) { return dim3((unsigned)((n + T - 1) / T)); };

    // degree -> dinv (in place)
    deg_init_kernel<<<g1(NN), blk, 0, stream>>>(dinv, NN);
    deg_acc_kernel<<<g1(E), blk, 0, stream>>>(dst, dinv, E);
    dinv_kernel<<<g1(NN), blk, 0, stream>>>(dinv, NN);

    // weight / input conversions
    cvt_wT_kernel<<<g1(DH * DH), blk, 0, stream>>>(W1, w1t);
    cvt_wT_kernel<<<g1(DH * DH), blk, 0, stream>>>(W2, w2t);
    cvt_f16_kernel<<<g1(total), blk, 0, stream>>>(x, hf16, total);

    const dim3 gemm_grid((unsigned)((NN / 16 + 7) / 8));         // 8 waves/block, 16 rows/wave
    const dim3 edge_grid((unsigned)(((size_t)E * 32 + T - 1) / T));

    // ---- layer 1 ----
    gemm_wmma_kernel<<<gemm_grid, blk, 0, stream>>>(hf16, w1t, hbuf, NN);
    agg_init_kernel<<<g1(total), blk, 0, stream>>>(hbuf, dinv, b1, abuf, total);
    agg_edges_kernel<<<edge_grid, blk, 0, stream>>>(hbuf, dinv, src, dst, abuf, E);
    relu_cvt_kernel<<<g1(total), blk, 0, stream>>>(abuf, hf16, total);

    // ---- layer 2 ----
    gemm_wmma_kernel<<<gemm_grid, blk, 0, stream>>>(hf16, w2t, hbuf, NN);
    agg_init_kernel<<<g1(total), blk, 0, stream>>>(hbuf, dinv, b2, abuf, total);
    agg_edges_kernel<<<edge_grid, blk, 0, stream>>>(hbuf, dinv, src, dst, abuf, E);

    // ---- head (relu fused) ----
    head_kernel<<<g1(NN), blk, 0, stream>>>(abuf, W3, b3, out, NN);
}
